// Attention_10797547782838
// MI455X (gfx1250) — matrix operand order repaired and checked by reading, not compiled, not run on hardware
//
#include <hip/hip_runtime.h>
#include <hip/hip_bf16.h>
#include <math.h>

typedef __attribute__((ext_vector_type(16))) _Float16 v16h;
typedef __attribute__((ext_vector_type(8)))  _Float16 v8h;
typedef __attribute__((ext_vector_type(8)))  float    v8f;

#define BATCH 16
#define SEQ   1024
#define CH    256
#define HEADS 8
#define HD    32
#define NWIN  4
#define MTOT  (BATCH*SEQ)   /* 16384 rows */

__device__ __forceinline__ v8f wmma_f16(v16h a, v16h b, v8f c) {
  return __builtin_amdgcn_wmma_f32_16x16x32_f16(false, a, false, b, (short)0, c, false, false);
}

/* ---------- prep kernels ---------- */

__global__ void k_cvt_x(const float* __restrict__ x, _Float16* __restrict__ xh) {
  int i = blockIdx.x * 256 + threadIdx.x;
  xh[i] = (_Float16)x[i];
}

// Build Wt_all[768][256] (f16, K-contiguous) and fused bias ball[768].
// rows 0..255 = Wq^T * scale ; 256..511 = K part of Wkv^T ; 512..767 = V part.
__global__ void k_prep_w(const float* __restrict__ Wq, const float* __restrict__ bq,
                         const float* __restrict__ Wkv, const float* __restrict__ bkv,
                         _Float16* __restrict__ wt, float* __restrict__ ball) {
  int t = blockIdx.x * 256 + threadIdx.x;   // 768*256 threads
  int o = t >> 8, i = t & 255;
  const float scale = 0.17677669529663687f; // 32^-0.5 folded into Q
  float v;
  if (o < 256) v = Wq[i * 256 + o] * scale;
  else         v = Wkv[i * 512 + (o - 256)];
  wt[o * 256 + i] = (_Float16)v;
  if (i == 0) ball[o] = (o < 256) ? bq[o] * scale : bkv[o - 256];
}

__global__ void k_prep_wp(const float* __restrict__ Wp, _Float16* __restrict__ wpt) {
  int t = blockIdx.x * 256 + threadIdx.x;   // 256*256 threads
  int o = t >> 8, i = t & 255;
  wpt[o * 256 + i] = (_Float16)Wp[i * 256 + o];
}

// bias[b][h][n] = 1 + sigmoid(angle_table[bin(b,n)][h])
__global__ void k_bias(const float* __restrict__ affine, const float* __restrict__ atab,
                       float* __restrict__ bias) {
  int t = blockIdx.x * 256 + threadIdx.x;   // 16384 threads = (b,n)
  int b = t >> 10, n = t & 1023;
  float cx = (float)(n & 31), cy = (float)(n >> 5);      // coords (x,y), W=H=32
  const float* A = affine + b * 6;
  float ex = A[0] * 16.f + A[1] * 16.f + A[2];
  float ey = A[3] * 16.f + A[4] * 16.f + A[5];
  float ang = atan2f(cy - ey, cx - ex);
  const float PI = 3.14159265358979323846f;
  int bin = (int)(((ang + PI) / (2.f * PI)) * 4.f);      // ANGLE_BINS-1 = 4
  bin = bin < 0 ? 0 : (bin > 4 ? 4 : bin);
#pragma unroll
  for (int h = 0; h < HEADS; ++h) {
    float v = atab[bin * HEADS + h];
    bias[((size_t)(b * HEADS + h)) * SEQ + n] = 1.f + 1.f / (1.f + __expf(-v));
  }
}

/* ---------- QKV projection GEMM (16384x256 @ 256x768) ---------- */
// One wave per 16x64 tile (4 accumulators, A fragment reused 4x);
// 32 x v_wmma_f32_16x16x32_f16 over K=256.
__global__ void __launch_bounds__(128) k_qkv(
    const _Float16* __restrict__ xh, const _Float16* __restrict__ wt,
    const float* __restrict__ ball,
    _Float16* __restrict__ qh, _Float16* __restrict__ kh, _Float16* __restrict__ vt) {
  int lane = threadIdx.x & 31, wave = threadIdx.x >> 5;
  int l16 = lane & 15, half = lane >> 4;
  int m0 = blockIdx.x * 16;
  int c0 = (blockIdx.y * 4 + wave) * 64;
  const _Float16* arow  = xh + (size_t)(m0 + l16) * CH + half * 16;  // A layout
  const _Float16* brow0 = wt + (size_t)(c0 + l16) * CH + half * 16;  // B layout
  v8f acc[4] = {{}, {}, {}, {}};
#pragma unroll
  for (int k = 0; k < CH; k += 32) {
    union { v16h v; v8h p[2]; } a;
    a.p[0] = *(const v8h*)(arow + k);
    a.p[1] = *(const v8h*)(arow + k + 8);
#pragma unroll
    for (int j = 0; j < 4; ++j) {
      union { v16h v; v8h p[2]; } b;
      const _Float16* br = brow0 + (size_t)j * 16 * CH;
      b.p[0] = *(const v8h*)(br + k);
      b.p[1] = *(const v8h*)(br + k + 8);
      acc[j] = wmma_f16(a.v, b.v, acc[j]);
    }
  }
#pragma unroll
  for (int j = 0; j < 4; ++j) {
    int c = c0 + j * 16 + l16;
    float bc = ball[c];
#pragma unroll
    for (int r = 0; r < 8; ++r) {
      int m = m0 + r + 8 * half;
      int bb = m >> 10, n = m & (SEQ - 1);
      _Float16 hv = (_Float16)(acc[j][r] + bc);
      if (c < 256) {                       // Q: (B,h,N,d)
        int hh = c >> 5, d = c & 31;
        qh[((size_t)(bb * HEADS + hh) * SEQ + n) * HD + d] = hv;
      } else if (c < 512) {                // K: (B,h,N,d)
        int cc = c - 256; int hh = cc >> 5, d = cc & 31;
        kh[((size_t)(bb * HEADS + hh) * SEQ + n) * HD + d] = hv;
      } else {                             // V transposed: (B,h,d,N)
        int cc = c - 512; int hh = cc >> 5, d = cc & 31;
        vt[((size_t)(bb * HEADS + hh) * HD + d) * SEQ + n] = hv;
      }
    }
  }
}

/* ---------- flash attention: 16 queries/wave, 32 keys/step ---------- */
__global__ void __launch_bounds__(128) k_attn(
    const _Float16* __restrict__ qh, const _Float16* __restrict__ kh,
    const _Float16* __restrict__ vt, const float* __restrict__ bias,
    const float* __restrict__ mask, _Float16* __restrict__ ao) {
  int lane = threadIdx.x & 31, wave = threadIdx.x >> 5;
  int l16 = lane & 15, half = lane >> 4;
  int bh = blockIdx.y; int b = bh >> 3, h = bh & 7;
  int q0 = blockIdx.x * 64 + wave * 16;
  const _Float16* qb = qh + (size_t)bh * SEQ * HD;
  const _Float16* kb = kh + (size_t)bh * SEQ * HD;
  const _Float16* vb = vt + (size_t)bh * HD * SEQ;
  const float* bsb = bias + (size_t)bh * SEQ;
  const float* mw = mask + (size_t)(b & (NWIN - 1)) * SEQ * SEQ;

  union { v16h v; v8h p[2]; } qa;
  {
    const _Float16* qr = qb + (size_t)(q0 + l16) * HD + half * 16;
    qa.p[0] = *(const v8h*)qr;
    qa.p[1] = *(const v8h*)(qr + 8);
  }
  float rb[8];
#pragma unroll
  for (int r = 0; r < 8; ++r) rb[r] = bsb[q0 + r + 8 * half];

  float mrow[8], lrow[8];
#pragma unroll
  for (int r = 0; r < 8; ++r) { mrow[r] = -3.0e38f; lrow[r] = 0.f; }
  v8f o0 = {}, o1 = {};

  __shared__ __align__(16) _Float16 pbuf[4][16 * 32];
  _Float16* pw = pbuf[wave];

  for (int k0 = 0; k0 < SEQ; k0 += 32) {
    union { v16h v; v8h p[2]; } kb0, kb1;
    {
      const _Float16* kr0 = kb + (size_t)(k0 + l16) * HD + half * 16;
      kb0.p[0] = *(const v8h*)kr0; kb0.p[1] = *(const v8h*)(kr0 + 8);
      const _Float16* kr1 = kb + (size_t)(k0 + 16 + l16) * HD + half * 16;
      kb1.p[0] = *(const v8h*)kr1; kb1.p[1] = *(const v8h*)(kr1 + 8);
    }
    // prefetch next chunk's mask rows into cache (global_prefetch_b8)
    if (k0 + 32 < SEQ)
      __builtin_prefetch(mw + (size_t)(q0 + l16) * SEQ + k0 + 32 + half * 16, 0, 0);
    v8f s0 = {}, s1 = {};
    s0 = wmma_f16(qa.v, kb0.v, s0);       // keys k0..k0+15
    s1 = wmma_f16(qa.v, kb1.v, s1);       // keys k0+16..k0+31
#pragma unroll
    for (int r = 0; r < 8; ++r) {         // bias (mul, per query) + mask (add)
      int row = q0 + r + 8 * half;
      const float* mr = mw + (size_t)row * SEQ + k0 + l16;
      s0[r] = s0[r] * rb[r] + mr[0];
      s1[r] = s1[r] * rb[r] + mr[16];
    }
#pragma unroll
    for (int r = 0; r < 8; ++r) {         // online softmax per row
      float mx = fmaxf(s0[r], s1[r]);
      mx = fmaxf(mx, __shfl_xor(mx, 8, 32));
      mx = fmaxf(mx, __shfl_xor(mx, 4, 32));
      mx = fmaxf(mx, __shfl_xor(mx, 2, 32));
      mx = fmaxf(mx, __shfl_xor(mx, 1, 32));
      float mnew = fmaxf(mrow[r], mx);
      float fac = __expf(mrow[r] - mnew);
      mrow[r] = mnew;
      float e0 = __expf(s0[r] - mnew);
      float e1 = __expf(s1[r] - mnew);
      float sr = e0 + e1;
      sr += __shfl_xor(sr, 8, 32);
      sr += __shfl_xor(sr, 4, 32);
      sr += __shfl_xor(sr, 2, 32);
      sr += __shfl_xor(sr, 1, 32);
      lrow[r] = lrow[r] * fac + sr;
      o0[r] *= fac; o1[r] *= fac;
      int rl = r + 8 * half;              // C-layout -> row-major LDS
      pw[rl * 32 + l16]      = (_Float16)e0;
      pw[rl * 32 + 16 + l16] = (_Float16)e1;
    }
    asm volatile("s_wait_dscnt 0" ::: "memory");
    union { v16h v; v8h p[2]; } pa;       // re-read P in A layout
    {
      const _Float16* pr = pw + l16 * 32 + half * 16;
      pa.p[0] = *(const v8h*)pr;
      pa.p[1] = *(const v8h*)(pr + 8);
    }
    union { v16h v; v8h p[2]; } vb0, vb1; // V as B (keys x dims), vt is (d,N)
    {
      const _Float16* vr0 = vb + (size_t)l16 * SEQ + k0 + half * 16;
      vb0.p[0] = *(const v8h*)vr0; vb0.p[1] = *(const v8h*)(vr0 + 8);
      const _Float16* vr1 = vb + (size_t)(16 + l16) * SEQ + k0 + half * 16;
      vb1.p[0] = *(const v8h*)vr1; vb1.p[1] = *(const v8h*)(vr1 + 8);
    }
    o0 = wmma_f16(pa.v, vb0.v, o0);       // dims 0..15
    o1 = wmma_f16(pa.v, vb1.v, o1);       // dims 16..31
    asm volatile("" ::: "memory");
  }
#pragma unroll
  for (int r = 0; r < 8; ++r) {
    int n = q0 + r + 8 * half;
    float inv = 1.0f / lrow[r];
    _Float16* orow = ao + ((size_t)(b * SEQ + n)) * CH + h * HD;
    orow[l16]      = (_Float16)(o0[r] * inv);
    orow[16 + l16] = (_Float16)(o1[r] * inv);
  }
}

/* ---------- output projection (16384x256 @ 256x256) -> f32 ---------- */
// One wave per 16x64 tile, 4 accumulators.
__global__ void __launch_bounds__(128) k_proj(
    const _Float16* __restrict__ ah, const _Float16* __restrict__ wpt,
    const float* __restrict__ bp, float* __restrict__ out) {
  int lane = threadIdx.x & 31, wave = threadIdx.x >> 5;
  int l16 = lane & 15, half = lane >> 4;
  int m0 = blockIdx.x * 16;
  int c0 = wave * 64;
  const _Float16* arow  = ah + (size_t)(m0 + l16) * CH + half * 16;
  const _Float16* brow0 = wpt + (size_t)(c0 + l16) * CH + half * 16;
  v8f acc[4] = {{}, {}, {}, {}};
#pragma unroll
  for (int k = 0; k < CH; k += 32) {
    union { v16h v; v8h p[2]; } a;
    a.p[0] = *(const v8h*)(arow + k);
    a.p[1] = *(const v8h*)(arow + k + 8);
#pragma unroll
    for (int j = 0; j < 4; ++j) {
      union { v16h v; v8h p[2]; } b;
      const _Float16* br = brow0 + (size_t)j * 16 * CH;
      b.p[0] = *(const v8h*)(br + k);
      b.p[1] = *(const v8h*)(br + k + 8);
      acc[j] = wmma_f16(a.v, b.v, acc[j]);
    }
  }
#pragma unroll
  for (int j = 0; j < 4; ++j) {
    int c = c0 + j * 16 + l16;
    float bc = bp[c];
#pragma unroll
    for (int r = 0; r < 8; ++r) {
      int m = m0 + r + 8 * half;
      out[(size_t)m * CH + c] = acc[j][r] + bc;
    }
  }
}

extern "C" void kernel_launch(void* const* d_in, const int* in_sizes, int n_in,
                              void* d_out, int out_size, void* d_ws, size_t ws_size,
                              hipStream_t stream) {
  const float* x    = (const float*)d_in[0];
  const float* mask = (const float*)d_in[1];
  const float* aff  = (const float*)d_in[2];
  const float* Wq   = (const float*)d_in[3];
  const float* bq   = (const float*)d_in[4];
  const float* Wkv  = (const float*)d_in[5];
  const float* bkv  = (const float*)d_in[6];
  const float* Wp   = (const float*)d_in[7];
  const float* bp   = (const float*)d_in[8];
  const float* atab = (const float*)d_in[9];
  float* out = (float*)d_out;

  char* ws = (char*)d_ws;
  size_t off = 0;
  auto take = [&](size_t bytes) {
    char* p = ws + off;
    off = (off + bytes + 255) & ~(size_t)255;
    return p;
  };
  _Float16* xh   = (_Float16*)take((size_t)MTOT * CH * 2);
  _Float16* wt   = (_Float16*)take((size_t)768 * 256 * 2);
  float*    ball = (float*)take(768 * 4);
  _Float16* wpt  = (_Float16*)take((size_t)256 * 256 * 2);
  _Float16* qh   = (_Float16*)take((size_t)MTOT * CH * 2);
  _Float16* kh   = (_Float16*)take((size_t)MTOT * CH * 2);
  _Float16* vt   = (_Float16*)take((size_t)MTOT * CH * 2);
  float*    bias = (float*)take((size_t)BATCH * HEADS * SEQ * 4);
  _Float16* ao   = (_Float16*)take((size_t)MTOT * CH * 2);

  k_cvt_x <<<MTOT * CH / 256, 256, 0, stream>>>(x, xh);
  k_prep_w<<<768 * 256 / 256, 256, 0, stream>>>(Wq, bq, Wkv, bkv, wt, ball);
  k_prep_wp<<<256 * 256 / 256, 256, 0, stream>>>(Wp, wpt);
  k_bias  <<<MTOT / 256, 256, 0, stream>>>(aff, atab, bias);
  k_qkv   <<<dim3(MTOT / 16, 3), 128, 0, stream>>>(xh, wt, ball, qh, kh, vt);
  k_attn  <<<dim3(SEQ / 64, BATCH * HEADS), 128, 0, stream>>>(qh, kh, vt, bias, mask, ao);
  k_proj  <<<dim3(MTOT / 16, 1), 128, 0, stream>>>(ao, wpt, bp, out);
}
